// DecoderRNN_13941463843628
// MI455X (gfx1250) — matrix operand order repaired and checked by reading, not compiled, not run on hardware
//
#include <hip/hip_runtime.h>
#include <hip/hip_bf16.h>
#include <math.h>

typedef __bf16 bf16;
typedef bf16  v8bf  __attribute__((ext_vector_type(8)));
typedef bf16  v16bf __attribute__((ext_vector_type(16)));
typedef float v8f   __attribute__((ext_vector_type(8)));

#define B_  64
#define S_  80
#define H_  512
#define E_  512
#define T_  30
#define V_  32000
#define TS_ 29   // T-1 decode steps

// ---------------------------------------------------------------------------
// Tiled bf16 WMMA GEMM:  out[m,n] = sum_k A[m,k] * W[n,k] + bias[n]
// A: (M,K) bf16 row-major, W: (N,K) bf16 row-major (i.e. X @ W^T).
// One wave -> 32x64 output tile: 2 A-fragments x 4 B-fragments
//             = 8x v_wmma_f32_16x16x32_bf16 per 32-wide K-step,
//             amortizing B loads over two M-tiles.
// Block = 128 threads = 4 waves arranged 2(M) x 2(N): block tile 64x128.
// grid = (N/128, ceil(M/64)).  N must be a multiple of 128, M of 32.
// Writes f32 (outF) and/or bf16 (outB), row stride ldOut.
// ---------------------------------------------------------------------------
__global__ __launch_bounds__(128)
void k_wmma_gemm_xwT(const bf16* __restrict__ A, const bf16* __restrict__ W,
                     const float* __restrict__ bias,
                     float* __restrict__ outF, bf16* __restrict__ outB,
                     int M, int N, int K, long long ldOut)
{
    const int lane = threadIdx.x & 31;
    const int wid  = threadIdx.x >> 5;
    const int half = lane >> 4;     // 0: lanes 0-15, 1: lanes 16-31
    const int l16  = lane & 15;
    const int m0 = blockIdx.y * 64  + (wid & 1) * 32;
    const int n0 = blockIdx.x * 128 + (wid >> 1) * 64;
    if (m0 >= M) return;

    v8f acc00 = {}, acc01 = {}, acc02 = {}, acc03 = {};
    v8f acc10 = {}, acc11 = {}, acc12 = {}, acc13 = {};

    const bf16* __restrict__ arow0 = A + (size_t)(m0 +      l16) * K;
    const bf16* __restrict__ arow1 = A + (size_t)(m0 + 16 + l16) * K;
    const bf16* __restrict__ w0 = W + (size_t)(n0 +  0 + l16) * K + half * 16;
    const bf16* __restrict__ w1 = W + (size_t)(n0 + 16 + l16) * K + half * 16;
    const bf16* __restrict__ w2 = W + (size_t)(n0 + 32 + l16) * K + half * 16;
    const bf16* __restrict__ w3 = W + (size_t)(n0 + 48 + l16) * K + half * 16;

    #pragma unroll 2
    for (int k0 = 0; k0 < K; k0 += 32) {
        // A fragments: lane m holds A[m][k0 + 16*half .. +15] (contiguous, matches B)
        v8bf a0lo = *(const v8bf*)(arow0 + k0 + half * 16);
        v8bf a0hi = *(const v8bf*)(arow0 + k0 + 8 + half * 16);
        v8bf a1lo = *(const v8bf*)(arow1 + k0 + half * 16);
        v8bf a1hi = *(const v8bf*)(arow1 + k0 + 8 + half * 16);
        v16bf a0 = __builtin_shufflevector(a0lo, a0hi,
                       0,1,2,3,4,5,6,7,8,9,10,11,12,13,14,15);
        v16bf a1 = __builtin_shufflevector(a1lo, a1hi,
                       0,1,2,3,4,5,6,7,8,9,10,11,12,13,14,15);
        // B fragments: lane n holds W[n][k0 + 16*half .. +15] (contiguous)
        v16bf b0 = *(const v16bf*)(w0 + k0);
        v16bf b1 = *(const v16bf*)(w1 + k0);
        v16bf b2 = *(const v16bf*)(w2 + k0);
        v16bf b3 = *(const v16bf*)(w3 + k0);

        acc00 = __builtin_amdgcn_wmma_f32_16x16x32_bf16(false, a0, false, b0, (short)0, acc00, false, false);
        acc01 = __builtin_amdgcn_wmma_f32_16x16x32_bf16(false, a0, false, b1, (short)0, acc01, false, false);
        acc02 = __builtin_amdgcn_wmma_f32_16x16x32_bf16(false, a0, false, b2, (short)0, acc02, false, false);
        acc03 = __builtin_amdgcn_wmma_f32_16x16x32_bf16(false, a0, false, b3, (short)0, acc03, false, false);
        acc10 = __builtin_amdgcn_wmma_f32_16x16x32_bf16(false, a1, false, b0, (short)0, acc10, false, false);
        acc11 = __builtin_amdgcn_wmma_f32_16x16x32_bf16(false, a1, false, b1, (short)0, acc11, false, false);
        acc12 = __builtin_amdgcn_wmma_f32_16x16x32_bf16(false, a1, false, b2, (short)0, acc12, false, false);
        acc13 = __builtin_amdgcn_wmma_f32_16x16x32_bf16(false, a1, false, b3, (short)0, acc13, false, false);
    }

    v8f accs[2][4] = {{acc00, acc01, acc02, acc03},
                      {acc10, acc11, acc12, acc13}};
    #pragma unroll
    for (int mt = 0; mt < 2; ++mt) {
        #pragma unroll
        for (int nt = 0; nt < 4; ++nt) {
            const int n = n0 + nt * 16 + l16;
            const float bv = bias ? bias[n] : 0.0f;
            #pragma unroll
            for (int r = 0; r < 8; ++r) {
                const int m = m0 + mt * 16 + r + 8 * half;   // C/D layout
                const long long off = (long long)m * ldOut + n;
                const float v = accs[mt][nt][r] + bv;
                if (outF) outF[off] = v;
                if (outB) outB[off] = (bf16)v;
            }
        }
    }
}

// ---------------------------------------------------------------------------
// Small helper kernels
// ---------------------------------------------------------------------------
__global__ void k_f32_to_bf16(const float* __restrict__ src,
                              bf16* __restrict__ dst, long long n)
{
    long long i = (long long)blockIdx.x * blockDim.x + threadIdx.x;
    if (i < n) dst[i] = (bf16)src[i];
}

// split W1 (H x 2H) -> W1_enc (H x H) and W1_hid (H x H), bf16
__global__ void k_split_W1(const float* __restrict__ W1,
                           bf16* __restrict__ W1e, bf16* __restrict__ W1h)
{
    int i = blockIdx.x * blockDim.x + threadIdx.x;      // H*H
    if (i >= H_ * H_) return;
    int h = i / H_, f = i % H_;
    W1e[i] = (bf16)W1[(size_t)h * (2 * H_) + f];
    W1h[i] = (bf16)W1[(size_t)h * (2 * H_) + H_ + f];
}

__global__ void k_zero_h(float* __restrict__ h, bf16* __restrict__ hb)
{
    int i = blockIdx.x * blockDim.x + threadIdx.x;
    if (i < B_ * H_) { h[i] = 0.0f; hb[i] = (bf16)0.0f; }
}

// x[b,s,:] = bf16( enc_part[b,s,:] + hW1[b,:] )
__global__ void k_make_x(const float* __restrict__ enc_part,
                         const float* __restrict__ hW1,
                         bf16* __restrict__ x)
{
    long long i = (long long)blockIdx.x * blockDim.x + threadIdx.x;
    if (i >= (long long)B_ * S_ * H_) return;
    int j  = (int)(i % H_);
    int b  = (int)(i / ((long long)S_ * H_));
    x[i] = (bf16)(enc_part[i] + hW1[b * H_ + j]);
}

// per-batch attention: scores = z . v_att, softmax over S, context = attn @ enc
__global__ __launch_bounds__(256)
void k_attention(const float* __restrict__ z, const float* __restrict__ v_att,
                 const float* __restrict__ enc, float* __restrict__ context)
{
    __shared__ float sc[S_];
    __shared__ float at[S_];
    const int b = blockIdx.x;
    const int lane = threadIdx.x & 31;
    const int wid  = threadIdx.x >> 5;      // 8 waves

    for (int s = wid; s < S_; s += 8) {
        const float* zp = z + ((size_t)b * S_ + s) * H_;
        float acc = 0.0f;
        for (int i = lane; i < H_; i += 32) acc += zp[i] * v_att[i];
        #pragma unroll
        for (int off = 16; off > 0; off >>= 1) acc += __shfl_xor(acc, off, 32);
        if (lane == 0) sc[s] = acc;
    }
    __syncthreads();
    if (threadIdx.x == 0) {
        float mx = -1e30f;
        for (int s = 0; s < S_; ++s) mx = fmaxf(mx, sc[s]);
        float sum = 0.0f;
        for (int s = 0; s < S_; ++s) { float e = __expf(sc[s] - mx); at[s] = e; sum += e; }
        float inv = 1.0f / sum;
        for (int s = 0; s < S_; ++s) at[s] *= inv;
    }
    __syncthreads();
    for (int f = threadIdx.x; f < H_; f += 256) {
        float acc = 0.0f;
        for (int s = 0; s < S_; ++s)
            acc += at[s] * enc[((size_t)b * S_ + s) * H_ + f];
        context[b * H_ + f] = acc;
    }
}

// gx[b, 0:E) = emb(targets[b,t]);  gx[b, E:E+H) = context[b,:]  (bf16)
__global__ void k_make_gx(const float* __restrict__ embedding,
                          const int* __restrict__ targets,
                          const float* __restrict__ context,
                          bf16* __restrict__ gx, int t)
{
    int i = blockIdx.x * blockDim.x + threadIdx.x;   // B*(E+H)
    if (i >= B_ * (E_ + H_)) return;
    int b = i / (E_ + H_), j = i % (E_ + H_);
    float v;
    if (j < E_) {
        int tok = targets[b * T_ + t];
        v = embedding[(size_t)tok * E_ + j];
    } else {
        v = context[b * H_ + (j - E_)];
    }
    gx[i] = (bf16)v;
}

// GRU gates: h_new = (1-z)*n + z*h
__global__ void k_gru_gate(const float* __restrict__ gi, const float* __restrict__ gh,
                           float* __restrict__ h, bf16* __restrict__ hb)
{
    int i = blockIdx.x * blockDim.x + threadIdx.x;   // B*H
    if (i >= B_ * H_) return;
    int b = i >> 9, j = i & (H_ - 1);
    const float* gib = gi + (size_t)b * 3 * H_;
    const float* ghb = gh + (size_t)b * 3 * H_;
    float r  = 1.0f / (1.0f + __expf(-(gib[j] + ghb[j])));
    float zz = 1.0f / (1.0f + __expf(-(gib[H_ + j] + ghb[H_ + j])));
    float nn = tanhf(gib[2 * H_ + j] + r * ghb[2 * H_ + j]);
    float hn = (1.0f - zz) * nn + zz * h[i];
    h[i]  = hn;
    hb[i] = (bf16)hn;
}

// argmax over V for row (b, t); preds written as float
__global__ __launch_bounds__(256)
void k_argmax(const float* __restrict__ logits_base, float* __restrict__ preds, int t)
{
    __shared__ float sv[256];
    __shared__ int   si[256];
    const int b = blockIdx.x;
    const float* row = logits_base + ((size_t)b * TS_ + t) * V_;
    float best = -1e30f; int bi = 0;
    for (int j = threadIdx.x; j < V_; j += 256) {
        float v = row[j];
        if (v > best) { best = v; bi = j; }
    }
    sv[threadIdx.x] = best; si[threadIdx.x] = bi;
    __syncthreads();
    for (int off = 128; off > 0; off >>= 1) {
        if (threadIdx.x < off) {
            float v2 = sv[threadIdx.x + off]; int i2 = si[threadIdx.x + off];
            if (v2 > sv[threadIdx.x] ||
                (v2 == sv[threadIdx.x] && i2 < si[threadIdx.x])) {
                sv[threadIdx.x] = v2; si[threadIdx.x] = i2;
            }
        }
        __syncthreads();
    }
    if (threadIdx.x == 0) preds[b * TS_ + t] = (float)si[0];
}

// ---------------------------------------------------------------------------
// Host launcher
// ---------------------------------------------------------------------------
extern "C" void kernel_launch(void* const* d_in, const int* in_sizes, int n_in,
                              void* d_out, int out_size, void* d_ws, size_t ws_size,
                              hipStream_t stream)
{
    const float* enc   = (const float*)d_in[0];   // (B,S,H)
    const int*   tgt   = (const int*)  d_in[1];   // (B,T)
    const float* emb   = (const float*)d_in[2];   // (V,E)
    const float* W1    = (const float*)d_in[3];   // (H,2H)
    const float* b1    = (const float*)d_in[4];
    const float* W2    = (const float*)d_in[5];   // (H,H)
    const float* b2    = (const float*)d_in[6];
    const float* W3    = (const float*)d_in[7];   // (H,H)
    const float* b3    = (const float*)d_in[8];
    const float* v_att = (const float*)d_in[9];   // (H)
    const float* W_ih  = (const float*)d_in[10];  // (3H, H+E)
    const float* b_ih  = (const float*)d_in[11];
    const float* W_hh  = (const float*)d_in[12];  // (3H, H)
    const float* b_hh  = (const float*)d_in[13];
    const float* W_out = (const float*)d_in[14];  // (V, H)
    const float* b_out = (const float*)d_in[15];

    float* out = (float*)d_out;                       // (B,TS,V) logits
    float* preds = out + (size_t)B_ * TS_ * V_;       // (B,TS) preds-as-float

    // workspace carve-out (256B aligned)
    char* p = (char*)d_ws;
    auto alloc = [&](size_t bytes) -> char* {
        char* r = p; p += (bytes + 255) & ~(size_t)255; return r;
    };
    bf16*  enc_b   = (bf16*) alloc((size_t)B_*S_*H_*2);
    bf16*  W1e_b   = (bf16*) alloc((size_t)H_*H_*2);
    bf16*  W1h_b   = (bf16*) alloc((size_t)H_*H_*2);
    bf16*  W2_b    = (bf16*) alloc((size_t)H_*H_*2);
    bf16*  W3_b    = (bf16*) alloc((size_t)H_*H_*2);
    bf16*  Wih_b   = (bf16*) alloc((size_t)3*H_*(H_+E_)*2);
    bf16*  Whh_b   = (bf16*) alloc((size_t)3*H_*H_*2);
    bf16*  Wout_b  = (bf16*) alloc((size_t)V_*H_*2);
    float* enc_part= (float*)alloc((size_t)B_*S_*H_*4);
    float* hW1     = (float*)alloc((size_t)B_*H_*4);
    bf16*  x_b     = (bf16*) alloc((size_t)B_*S_*H_*2);
    bf16*  y_b     = (bf16*) alloc((size_t)B_*S_*H_*2);
    float* z_f     = (float*)alloc((size_t)B_*S_*H_*4);
    float* ctx     = (float*)alloc((size_t)B_*H_*4);
    bf16*  gx_b    = (bf16*) alloc((size_t)B_*(H_+E_)*2);
    float* gi      = (float*)alloc((size_t)B_*3*H_*4);
    float* gh      = (float*)alloc((size_t)B_*3*H_*4);
    float* h_f     = (float*)alloc((size_t)B_*H_*4);
    bf16*  h_b     = (bf16*) alloc((size_t)B_*H_*2);
    (void)ws_size; (void)in_sizes; (void)n_in; (void)out_size;

    const int TPB = 256;
    auto blocks = [](long long n, int tpb) { return (unsigned)((n + tpb - 1) / tpb); };

    // --- one-time conversions ---
    k_split_W1<<<blocks((long long)H_*H_, TPB), TPB, 0, stream>>>(W1, W1e_b, W1h_b);
    k_f32_to_bf16<<<blocks((long long)H_*H_, TPB), TPB, 0, stream>>>(W2,    W2_b,  (long long)H_*H_);
    k_f32_to_bf16<<<blocks((long long)H_*H_, TPB), TPB, 0, stream>>>(W3,    W3_b,  (long long)H_*H_);
    k_f32_to_bf16<<<blocks((long long)3*H_*(H_+E_), TPB), TPB, 0, stream>>>(W_ih, Wih_b, (long long)3*H_*(H_+E_));
    k_f32_to_bf16<<<blocks((long long)3*H_*H_, TPB), TPB, 0, stream>>>(W_hh,  Whh_b, (long long)3*H_*H_);
    k_f32_to_bf16<<<blocks((long long)V_*H_, TPB), TPB, 0, stream>>>(W_out,  Wout_b,(long long)V_*H_);
    k_f32_to_bf16<<<blocks((long long)B_*S_*H_, TPB), TPB, 0, stream>>>(enc,  enc_b, (long long)B_*S_*H_);
    k_zero_h<<<blocks(B_*H_, TPB), TPB, 0, stream>>>(h_f, h_b);

    // enc_part = enc @ W1_enc^T + b1   (M=B*S, N=H, K=H)
    {
        dim3 g(H_/128, (B_*S_ + 63)/64);
        k_wmma_gemm_xwT<<<g, 128, 0, stream>>>(enc_b, W1e_b, b1,
                                               enc_part, nullptr,
                                               B_*S_, H_, H_, (long long)H_);
    }

    // --- 29 sequential decode steps ---
    for (int t = 0; t < TS_; ++t) {
        // hW1 = h @ W1_hid^T   (M=B, N=H, K=H)
        {
            dim3 g(H_/128, 1);
            k_wmma_gemm_xwT<<<g, 128, 0, stream>>>(h_b, W1h_b, nullptr,
                                                   hW1, nullptr,
                                                   B_, H_, H_, (long long)H_);
        }
        // x = bf16(enc_part + broadcast(hW1))
        k_make_x<<<blocks((long long)B_*S_*H_, TPB), TPB, 0, stream>>>(enc_part, hW1, x_b);
        // y = x @ W2^T + b2  (bf16 out)
        {
            dim3 g(H_/128, (B_*S_ + 63)/64);
            k_wmma_gemm_xwT<<<g, 128, 0, stream>>>(x_b, W2_b, b2,
                                                   nullptr, y_b,
                                                   B_*S_, H_, H_, (long long)H_);
        }
        // z = y @ W3^T + b3  (f32 out)
        {
            dim3 g(H_/128, (B_*S_ + 63)/64);
            k_wmma_gemm_xwT<<<g, 128, 0, stream>>>(y_b, W3_b, b3,
                                                   z_f, nullptr,
                                                   B_*S_, H_, H_, (long long)H_);
        }
        // attention softmax + context
        k_attention<<<B_, 256, 0, stream>>>(z_f, v_att, enc, ctx);
        // gx = [emb_t, context] (bf16)
        k_make_gx<<<blocks(B_*(H_+E_), TPB), TPB, 0, stream>>>(emb, tgt, ctx, gx_b, t);
        // gi = gx @ W_ih^T + b_ih  (M=B, N=3H, K=H+E)
        {
            dim3 g((3*H_)/128, 1);
            k_wmma_gemm_xwT<<<g, 128, 0, stream>>>(gx_b, Wih_b, b_ih,
                                                   gi, nullptr,
                                                   B_, 3*H_, H_+E_, (long long)(3*H_));
        }
        // gh = h @ W_hh^T + b_hh  (M=B, N=3H, K=H)
        {
            dim3 g((3*H_)/128, 1);
            k_wmma_gemm_xwT<<<g, 128, 0, stream>>>(h_b, Whh_b, b_hh,
                                                   gh, nullptr,
                                                   B_, 3*H_, H_, (long long)(3*H_));
        }
        // GRU gates -> h (f32 + bf16)
        k_gru_gate<<<blocks(B_*H_, TPB), TPB, 0, stream>>>(gi, gh, h_f, h_b);
        // logits[b,t,:] = h_new @ W_out^T + b_out  (M=B, N=V, K=H)
        {
            dim3 g(V_/128, 1);
            k_wmma_gemm_xwT<<<g, 128, 0, stream>>>(h_b, Wout_b, b_out,
                                                   out + (size_t)t * V_, nullptr,
                                                   B_, V_, H_, (long long)TS_ * V_);
        }
        // argmax -> preds
        k_argmax<<<B_, 256, 0, stream>>>(out, preds, t);
    }
}
